// TransformerEncoderBlock_34437047779658
// MI455X (gfx1250) — matrix operand order repaired and checked by reading, not compiled, not run on hardware
//
#include <hip/hip_runtime.h>
#include <cstdint>
#include <cstddef>

typedef __bf16 bf16;
typedef __attribute__((ext_vector_type(16))) __bf16 v16bf;
typedef __attribute__((ext_vector_type(8)))  float  v8f;

#define EMB   768
#define NH    8
#define HD    96
#define HID   3072
#define SEQ   2048
#define BATCH 4
#define TOK   (BATCH * SEQ)

static __device__ __forceinline__ v8f wmma_bf16(v16bf a, v16bf b, v8f c) {
  // D = A(16x32 bf16) * B(32x16 bf16) + C(16x16 f32)
  return __builtin_amdgcn_wmma_f32_16x16x32_bf16(false, a, false, b, (short)0, c,
                                                 false, false);
}

// ---------------------------------------------------------------------------
// 16-lane butterfly reductions in pure VALU via DPP16 (no LDS, no waits).
// Steps: quad_perm xor1 (0xB1), quad_perm xor2 (0x4E), row_half_mirror (0x141),
// row_mirror (0x140). DPP rows are 16 lanes, so lanes 0-15 and 16-31 reduce
// independently -- matching the WMMA C/D layout where each output row lives
// entirely inside one 16-lane half.
// ---------------------------------------------------------------------------
static __device__ __forceinline__ float dpp_max16(float x) {
  int t;
  t = __builtin_amdgcn_mov_dpp(__float_as_int(x), 0xB1,  0xF, 0xF, true);
  x = fmaxf(x, __int_as_float(t));
  t = __builtin_amdgcn_mov_dpp(__float_as_int(x), 0x4E,  0xF, 0xF, true);
  x = fmaxf(x, __int_as_float(t));
  t = __builtin_amdgcn_mov_dpp(__float_as_int(x), 0x141, 0xF, 0xF, true);
  x = fmaxf(x, __int_as_float(t));
  t = __builtin_amdgcn_mov_dpp(__float_as_int(x), 0x140, 0xF, 0xF, true);
  x = fmaxf(x, __int_as_float(t));
  return x;
}

static __device__ __forceinline__ float dpp_sum16(float x) {
  int t;
  t = __builtin_amdgcn_mov_dpp(__float_as_int(x), 0xB1,  0xF, 0xF, true);
  x += __int_as_float(t);
  t = __builtin_amdgcn_mov_dpp(__float_as_int(x), 0x4E,  0xF, 0xF, true);
  x += __int_as_float(t);
  t = __builtin_amdgcn_mov_dpp(__float_as_int(x), 0x141, 0xF, 0xF, true);
  x += __int_as_float(t);
  t = __builtin_amdgcn_mov_dpp(__float_as_int(x), 0x140, 0xF, 0xF, true);
  x += __int_as_float(t);
  return x;
}

// ---------------------------------------------------------------------------
// Pack a row-major fp32 weight W[K,N] into bf16 B-fragment order:
// fragment (kt, nt) covers K-rows [32*kt,32*kt+32), N-cols [16*nt,16*nt+16).
// This code's contiguous K order (16-bit B 32x16): lane = 16*((k%32)/16) + (n%16),
// element e = k%16 (16 contiguous K per lane).
// Packed addr: ((kt*(N/16)+nt)*32 + lane)*16 + e  -> one coalesced 32B/lane load.
// ---------------------------------------------------------------------------
__global__ void pack_w_kernel(const float* __restrict__ W, bf16* __restrict__ dst,
                              int K, int N) {
  int idx = blockIdx.x * blockDim.x + threadIdx.x;
  if (idx >= K * N) return;
  int k = idx / N, n = idx % N;
  int kt = k >> 5;
  int lane = (((k >> 4) & 1) << 4) + (n & 15);
  int e = k & 15;
  size_t frag = (size_t)kt * (N >> 4) + (n >> 4);
  dst[(frag * 32 + lane) * 16 + e] = (bf16)W[idx];
}

// ---------------------------------------------------------------------------
// LayerNorm over 768 cols, fp32 in -> bf16 out. One 256-thread block per row.
// ---------------------------------------------------------------------------
__global__ void ln_kernel(const float* __restrict__ x, const float* __restrict__ g,
                          const float* __restrict__ b, bf16* __restrict__ out) {
  int row = blockIdx.x;
  int tid = threadIdx.x;
  const float* xr = x + (size_t)row * EMB;
  float v[3], s = 0.f, s2 = 0.f;
#pragma unroll
  for (int i = 0; i < 3; ++i) { v[i] = xr[tid + 256 * i]; s += v[i]; s2 += v[i] * v[i]; }
  s  = dpp_sum16(s);
  s2 = dpp_sum16(s2);
  s  += __shfl_xor(s,  16, 32);
  s2 += __shfl_xor(s2, 16, 32);
  __shared__ float ssum[8], ssq[8];
  if ((tid & 31) == 0) { ssum[tid >> 5] = s; ssq[tid >> 5] = s2; }
  __syncthreads();
  float ts = 0.f, tq = 0.f;
#pragma unroll
  for (int w = 0; w < 8; ++w) { ts += ssum[w]; tq += ssq[w]; }
  float mu  = ts * (1.f / EMB);
  float var = tq * (1.f / EMB) - mu * mu;
  float rs  = rsqrtf(var + 1e-5f);
  bf16* orow = out + (size_t)row * EMB;
#pragma unroll
  for (int i = 0; i < 3; ++i) {
    int c = tid + 256 * i;
    orow[c] = (bf16)((v[i] - mu) * rs * g[c] + b[c]);
  }
}

// ---------------------------------------------------------------------------
// WMMA GEMM: C[M,N] = A[M,K](row-major bf16) * Wpk(K,N prepacked fragments).
// One wave computes a 32x64 tile: 2 M-tiles x 4 N-tiles = 8 WMMAs per K-step.
// Each A fragment is reused across 4 N-tiles, each B fragment across 2
// M-tiles -> ~21 FLOP per L2 byte, ~110 VGPRs (2 waves/SIMD occupancy).
// Epilogues:
//   EPI_QKV      : +bias, scatter bf16 into packed Q/K/V fragment layouts
//   EPI_BIAS_RES : +bias +resid(f32,[M,N]) -> f32 out
//   EPI_GELU     : +bias, exact gelu       -> bf16 out (row-major)
// ---------------------------------------------------------------------------
enum { EPI_QKV = 0, EPI_BIAS_RES = 1, EPI_GELU = 2 };

template <int EPI>
__global__ __launch_bounds__(256, 2)
void gemm_bf16(const bf16* __restrict__ A, const bf16* __restrict__ Wpk,
               int N, int K,
               const float* __restrict__ bias,
               const float* __restrict__ resid,
               float* __restrict__ outf, bf16* __restrict__ outb,
               bf16* __restrict__ Qpk, bf16* __restrict__ Kpk,
               bf16* __restrict__ Vpk) {
  int gw   = (blockIdx.x * blockDim.x + threadIdx.x) >> 5;
  int lane = threadIdx.x & 31;
  int wN = N >> 6;
  int mt = gw / wN, nw = gw % wN;
  int m0 = mt << 5, n0 = nw << 6;
  int ntiles = N >> 4;
  int half = lane >> 4;
  int mrow = lane & 15;

  v8f z = {0.f, 0.f, 0.f, 0.f, 0.f, 0.f, 0.f, 0.f};
  v8f acc[2][4] = {{z, z, z, z}, {z, z, z, z}};

  const bf16* Arow0 = A + (size_t)(m0 + mrow) * K;
  const bf16* Arow1 = Arow0 + (size_t)16 * K;
  for (int kt = 0; kt < (K >> 5); ++kt) {
    // A fragments (16x32 bf16): lane holds row mrow; k = 16*half + 2*vin + p
    union { v16bf v; uint32_t u[8]; } af0, af1;
    const bf16* ap0 = Arow0 + kt * 32 + half * 16;
    const bf16* ap1 = Arow1 + kt * 32 + half * 16;
#pragma unroll
    for (int j = 0; j < 8; ++j) {
      int k = j << 1;
      af0.u[j] = *(const uint32_t*)(ap0 + k);
      af1.u[j] = *(const uint32_t*)(ap1 + k);
    }
#pragma unroll
    for (int t = 0; t < 4; ++t) {
      v16bf bfr = *(const v16bf*)(Wpk + ((size_t)kt * ntiles + (n0 >> 4) + t) * 512 +
                                  lane * 16);
      acc[0][t] = wmma_bf16(af0.v, bfr, acc[0][t]);
      acc[1][t] = wmma_bf16(af1.v, bfr, acc[1][t]);
    }
  }

#pragma unroll
  for (int mi = 0; mi < 2; ++mi) {
#pragma unroll
    for (int t = 0; t < 4; ++t) {
#pragma unroll
      for (int r = 0; r < 8; ++r) {
        float vv = acc[mi][t][r];
        int m = m0 + mi * 16 + r + 8 * half;   // C/D layout: row = r + 8*(lane>=16)
        int c = n0 + (t << 4) + mrow;          // col = lane&15 within tile
        vv += bias[c];
        if constexpr (EPI == EPI_QKV) {
          // qkv column layout: c = head*288 + dd*3 + which
          int which = c % 3;
          int dd    = (c / 3) % HD;
          int head  = c / (3 * HD);
          int bb = m >> 11;                    // batch = m / 2048
          int n  = m & 2047;                   // position in sequence
          int bh = bb * NH + head;
          bf16 hv = (bf16)vv;
          if (which == 0) {                    // Q -> A-fragment layout [bh][qt][kc]
            int qt = n >> 4, mr = n & 15;
            int kc = dd / 32, kin = dd % 32;
            int lane2 = (((kin >> 4) & 1) << 4) + mr;
            int e = kin & 15;
            Qpk[(((size_t)bh * 128 + qt) * 3 + kc) * 512 + lane2 * 16 + e] = hv;
          } else if (which == 1) {             // K^T -> B-fragment layout [bh][nt][kc]
            int ntt = n >> 4;
            int kc = dd >> 5;
            int lane2 = (((dd >> 4) & 1) << 4) + (n & 15);
            int e = dd & 15;
            Kpk[(((size_t)bh * 128 + ntt) * 3 + kc) * 512 + lane2 * 16 + e] = hv;
          } else {                             // V -> B-fragment layout [bh][kt][dt]
            int kt2 = n >> 5;
            int tt  = dd >> 4;
            int lane2 = (((n >> 4) & 1) << 4) + (dd & 15);
            int e = n & 15;
            Vpk[(((size_t)bh * 64 + kt2) * 6 + tt) * 512 + lane2 * 16 + e] = hv;
          }
        } else if constexpr (EPI == EPI_BIAS_RES) {
          vv += resid[(size_t)m * N + c];
          outf[(size_t)m * N + c] = vv;
        } else {                               // EPI_GELU (exact)
          float gg = 0.5f * vv * (1.f + erff(vv * 0.70710678118654752f));
          outb[(size_t)m * N + c] = (bf16)gg;
        }
      }
    }
  }
}

// ---------------------------------------------------------------------------
// Flash attention, one wave per 16-query block per (batch, head).
// Streams 32 keys/iter: 6 WMMAs for S = Q K^T, online softmax with per-row
// stats reduced via DPP16 butterflies (rows live in 16-lane halves), P
// re-shaped C-layout -> A-layout through per-wave LDS, 6 WMMAs for O += P V.
// The 8 waves of a block handle consecutive q-blocks of the SAME (b,h), so
// their K/V fragment streams share cache lines.
// Final: O / rowsum / sqrt(768)  (reference's post-softmax scaling quirk).
// ---------------------------------------------------------------------------
__global__ __launch_bounds__(256, 1)
void attn_kernel(const bf16* __restrict__ Qpk, const bf16* __restrict__ Kpk,
                 const bf16* __restrict__ Vpk, bf16* __restrict__ attout) {
  __shared__ alignas(16) bf16 psh[8][16 * 32];
  int wid = threadIdx.x >> 5, lane = threadIdx.x & 31;
  int gw = blockIdx.x * 8 + wid;           // 0 .. 4095
  int bh = gw >> 7;
  int qt = gw & 127;
  int half = lane >> 4, nlane = lane & 15;

  const bf16* Qb = Qpk + ((size_t)bh * 128 + qt) * 3 * 512 + lane * 16;
  v16bf qf0 = *(const v16bf*)(Qb);
  v16bf qf1 = *(const v16bf*)(Qb + 512);
  v16bf qf2 = *(const v16bf*)(Qb + 1024);

  float mrun[8], lrun[8];
#pragma unroll
  for (int r = 0; r < 8; ++r) { mrun[r] = -3.0e38f; lrun[r] = 0.f; }
  v8f z = {0.f, 0.f, 0.f, 0.f, 0.f, 0.f, 0.f, 0.f};
  v8f acc[6] = {z, z, z, z, z, z};

  bf16* P = &psh[wid][0];
  const bf16* Kb = Kpk + (size_t)bh * 128 * 3 * 512 + lane * 16;
  const bf16* Vb = Vpk + (size_t)bh * 64 * 6 * 512 + lane * 16;

  for (int kb = 0; kb < 64; ++kb) {
    const bf16* k0 = Kb + (size_t)(2 * kb) * 3 * 512;
    const bf16* k1 = k0 + 3 * 512;
    v8f s0 = z, s1 = z;
    s0 = wmma_bf16(qf0, *(const v16bf*)(k0),        s0);
    s0 = wmma_bf16(qf1, *(const v16bf*)(k0 + 512),  s0);
    s0 = wmma_bf16(qf2, *(const v16bf*)(k0 + 1024), s0);
    s1 = wmma_bf16(qf0, *(const v16bf*)(k1),        s1);
    s1 = wmma_bf16(qf1, *(const v16bf*)(k1 + 512),  s1);
    s1 = wmma_bf16(qf2, *(const v16bf*)(k1 + 1024), s1);

#pragma unroll
    for (int r = 0; r < 8; ++r) {
      float a0 = s0[r], a1 = s1[r];
      float mx = dpp_max16(fmaxf(a0, a1));         // row max, VALU only
      float nm = fmaxf(mrun[r], mx);
      float sc = __expf(mrun[r] - nm);
      float p0 = __expf(a0 - nm), p1 = __expf(a1 - nm);
      float rsum = dpp_sum16(p0 + p1);             // row sum, VALU only
      lrun[r] = lrun[r] * sc + rsum;
      mrun[r] = nm;
#pragma unroll
      for (int t = 0; t < 6; ++t) acc[t][r] *= sc;
      int mm = r + 8 * half;
      P[mm * 32 + nlane]      = (bf16)p0;
      P[mm * 32 + 16 + nlane] = (bf16)p1;
    }
    asm volatile("s_wait_dscnt 0" ::: "memory");   // LDS store->load, wave-private
    union { v16bf v; uint32_t u[8]; } pf;
    const bf16* pr = P + nlane * 32 + half * 16;   // A-frag: row = lane&15
#pragma unroll
    for (int j = 0; j < 8; ++j) {
      int k = j << 1;
      pf.u[j] = *(const uint32_t*)(pr + k);
    }
    asm volatile("s_wait_dscnt 0" ::: "memory");   // WAR before next iter's stores
    const bf16* vbase = Vb + (size_t)kb * 6 * 512;
#pragma unroll
    for (int t = 0; t < 6; ++t)
      acc[t] = wmma_bf16(pf.v, *(const v16bf*)(vbase + t * 512), acc[t]);
  }

  const float inv = 0.03608439182435161f;          // 1/sqrt(768)
  int bb = bh >> 3, hh = bh & 7;
#pragma unroll
  for (int t = 0; t < 6; ++t) {
#pragma unroll
    for (int r = 0; r < 8; ++r) {
      int row = bb * SEQ + qt * 16 + r + 8 * half;
      int col = hh * HD + t * 16 + nlane;
      attout[(size_t)row * EMB + col] = (bf16)(acc[t][r] / lrun[r] * inv);
    }
  }
}

// ---------------------------------------------------------------------------
extern "C" void kernel_launch(void* const* d_in, const int* in_sizes, int n_in,
                              void* d_out, int out_size, void* d_ws, size_t ws_size,
                              hipStream_t stream) {
  (void)in_sizes; (void)n_in; (void)out_size; (void)ws_size;
  const float* x      = (const float*)d_in[0];
  const float* ln1_g  = (const float*)d_in[1];
  const float* ln1_b  = (const float*)d_in[2];
  const float* qkv_w  = (const float*)d_in[3];
  const float* qkv_b  = (const float*)d_in[4];
  const float* proj_w = (const float*)d_in[5];
  const float* proj_b = (const float*)d_in[6];
  const float* ln2_g  = (const float*)d_in[7];
  const float* ln2_b  = (const float*)d_in[8];
  const float* fc1_w  = (const float*)d_in[9];
  const float* fc1_b  = (const float*)d_in[10];
  const float* fc2_w  = (const float*)d_in[11];
  const float* fc2_b  = (const float*)d_in[12];
  float* out = (float*)d_out;

  char* ws = (char*)d_ws;
  size_t off = 0;
  auto take = [&](size_t bytes) -> char* {
    char* p = ws + off;
    off = (off + bytes + 255) & ~(size_t)255;
    return p;
  };
  bf16* qkvw_pk = (bf16*)take((size_t)EMB * 3 * EMB * 2);
  bf16* projw_pk = (bf16*)take((size_t)EMB * EMB * 2);
  bf16* fc1w_pk = (bf16*)take((size_t)EMB * HID * 2);
  bf16* fc2w_pk = (bf16*)take((size_t)HID * EMB * 2);
  bf16* hbf    = (bf16*)take((size_t)TOK * EMB * 2);   // LN1 out, reused for LN2 out
  bf16* Qpk    = (bf16*)take((size_t)TOK * EMB * 2);
  bf16* Kpk    = (bf16*)take((size_t)TOK * EMB * 2);
  bf16* Vpk    = (bf16*)take((size_t)TOK * EMB * 2);
  bf16* attout = (bf16*)take((size_t)TOK * EMB * 2);
  float* x1    = (float*)take((size_t)TOK * EMB * 4);
  bf16* h1     = (bf16*)take((size_t)TOK * HID * 2);

  // 1) weight prepack into WMMA B-fragment order
  pack_w_kernel<<<(EMB * 3 * EMB + 255) / 256, 256, 0, stream>>>(qkv_w, qkvw_pk, EMB, 3 * EMB);
  pack_w_kernel<<<(EMB * EMB + 255) / 256, 256, 0, stream>>>(proj_w, projw_pk, EMB, EMB);
  pack_w_kernel<<<(EMB * HID + 255) / 256, 256, 0, stream>>>(fc1_w, fc1w_pk, EMB, HID);
  pack_w_kernel<<<(HID * EMB + 255) / 256, 256, 0, stream>>>(fc2_w, fc2w_pk, HID, EMB);

  // 2) LN1
  ln_kernel<<<TOK, 256, 0, stream>>>(x, ln1_g, ln1_b, hbf);

  // 3) QKV GEMM, epilogue scatters into packed Q/K/V fragments
  {
    int N = 3 * EMB, K = EMB;
    int waves = (TOK / 32) * (N / 64);
    gemm_bf16<EPI_QKV><<<waves / 8, 256, 0, stream>>>(
        hbf, qkvw_pk, N, K, qkv_b, nullptr, nullptr, nullptr, Qpk, Kpk, Vpk);
  }

  // 4) flash attention (4096 waves, 8 per block)
  attn_kernel<<<(BATCH * NH * (SEQ / 16)) / 8, 256, 0, stream>>>(Qpk, Kpk, Vpk, attout);

  // 5) proj GEMM + bias + residual(x) -> x1 (f32)
  {
    int N = EMB, K = EMB;
    int waves = (TOK / 32) * (N / 64);
    gemm_bf16<EPI_BIAS_RES><<<waves / 8, 256, 0, stream>>>(
        attout, projw_pk, N, K, proj_b, x, x1, nullptr, nullptr, nullptr, nullptr);
  }

  // 6) LN2
  ln_kernel<<<TOK, 256, 0, stream>>>(x1, ln2_g, ln2_b, hbf);

  // 7) FC1 GEMM + bias + exact GELU -> h1 (bf16)
  {
    int N = HID, K = EMB;
    int waves = (TOK / 32) * (N / 64);
    gemm_bf16<EPI_GELU><<<waves / 8, 256, 0, stream>>>(
        hbf, fc1w_pk, N, K, fc1_b, nullptr, nullptr, h1, nullptr, nullptr, nullptr);
  }

  // 8) FC2 GEMM + bias + residual(x1) -> out (f32)
  {
    int N = EMB, K = HID;
    int waves = (TOK / 32) * (N / 64);
    gemm_bf16<EPI_BIAS_RES><<<waves / 8, 256, 0, stream>>>(
        h1, fc2w_pk, N, K, fc2_b, x1, out, nullptr, nullptr, nullptr, nullptr);
  }
}
